// NonLocal_EmbeddedGaussian_84834194031021
// MI455X (gfx1250) — matrix operand order repaired and checked by reading, not compiled, not run on hardware
//
#include <hip/hip_runtime.h>
#include <hip/hip_bf16.h>

// ---------------------------------------------------------------------------
// NonLocal Embedded-Gaussian block on MI455X (gfx1250, wave32, WMMA bf16).
// All 6 GEMMs run on v_wmma_f32_16x16x32_bf16. Softmax fused into score GEMM.
// Each wave computes a 16x128 output strip (8 WMMA tiles) reusing the A
// fragment across all 8 B tiles per k-step.
// ---------------------------------------------------------------------------

typedef __bf16 bf16;
typedef __attribute__((ext_vector_type(16))) __bf16 bf16x16;
typedef __attribute__((ext_vector_type(8)))  float  floatx8;

#define NB    8
#define CINC  1024
#define HH    64
#define WWID  64
#define NP    4096   // H*W
#define MP    1024   // (H/2)*(W/2)
#define DIM   512
#define COUT  1024
#define BN_EPS 1e-5f

// ---------------- WMMA helpers (ISA 7.12.2 lane layouts) -------------------

__device__ __forceinline__ floatx8 wmma_bf16(bf16x16 a, bf16x16 b, floatx8 c) {
  // 8 args: (neg_a, A, neg_b, B, c_mod, C, reuse_a, reuse_b)
  return __builtin_amdgcn_wmma_f32_16x16x32_bf16(false, a, false, b, (short)0, c,
                                                 false, false);
}

// A-matrix 16x32 bf16 fragment from row-major [rows][ld] (K contiguous).
// lane<16: M=lane, K pairs {0,2,4,6} and {16,18,20,22};
// lane>=16: M=lane-16, K pairs {8..14} and {24..30}.
__device__ __forceinline__ bf16x16 load_a_frag(const bf16* __restrict__ base,
                                               int ld, int row0, int k0) {
  const int lane = threadIdx.x & 31;
  const int half = lane >> 4;
  const int r    = lane & 15;
  const bf16* p = base + (size_t)(row0 + r) * ld + k0;
  bf16x16 f;
#pragma unroll
  for (int j = 0; j < 8; ++j) {
    const int kk = (j < 4) ? (8 * half + 2 * j) : (16 + 8 * half + 2 * (j - 4));
    f[2 * j]     = p[kk];
    f[2 * j + 1] = p[kk + 1];
  }
  return f;
}

// B-matrix 32x16 bf16 fragment from row-major Bt[N][K] (K contiguous).
// lane&15 = N; lane<16: K {0..7,16..23}, lane>=16: K {8..15,24..31} (same as A).
__device__ __forceinline__ bf16x16 load_b_frag(const bf16* __restrict__ base,
                                               int ld, int col0, int k0) {
  const int lane = threadIdx.x & 31;
  const int half = lane >> 4;
  const int n    = lane & 15;
  const bf16* p = base + (size_t)(col0 + n) * ld + k0 + 8 * half;
  bf16x16 f;
#pragma unroll
  for (int j = 0; j < 8; ++j) {
    const int kk = (j < 4) ? (2 * j) : (16 + 2 * (j - 4));
    f[2 * j]     = p[kk];
    f[2 * j + 1] = p[kk + 1];
  }
  return f;
}

// ---------------- generic NT GEMM: D[i][j] = sum_k A[i][k]*B[j][k] ----------
// A,B bf16 row-major with K contiguous. Per-batch strides (0 => broadcast
// weights). Output either bf16 (Dh) or f32 (Df). Optional per-row / per-col
// bias. Block = 128 threads = 4 waves; each wave owns a 16x128 strip
// (8 tiles), A fragment loaded once per k-step and reused 8x.
// Requires J % 512 == 0 (block covers 512 columns).
__global__ void gemm_nt(const bf16* __restrict__ A, size_t strideA, int ldA,
                        const bf16* __restrict__ B, size_t strideB, int ldB,
                        bf16* __restrict__ Dh, float* __restrict__ Df,
                        size_t strideD, int ldD,
                        const float* __restrict__ bias_i,
                        const float* __restrict__ bias_j, int K) {
  const int w     = threadIdx.x >> 5;
  const int i0    = blockIdx.x * 16;
  const int jbase = blockIdx.y * 512 + w * 128;
  const int z     = blockIdx.z;
  const bf16* Ab = A + strideA * z;
  const bf16* Bb = B + strideB * z;

  floatx8 acc[8];
#pragma unroll
  for (int t = 0; t < 8; ++t) acc[t] = (floatx8)(0.0f);

  for (int k = 0; k < K; k += 32) {
    bf16x16 af = load_a_frag(Ab, ldA, i0, k);
#pragma unroll
    for (int t = 0; t < 8; ++t) {
      bf16x16 bfr = load_b_frag(Bb, ldB, jbase + t * 16, k);
      acc[t] = wmma_bf16(af, bfr, acc[t]);
    }
  }

  const int lane = threadIdx.x & 31;
  const int half = lane >> 4;
  const int nl   = lane & 15;

  float bi[8];
#pragma unroll
  for (int r = 0; r < 8; ++r)
    bi[r] = bias_i ? bias_i[i0 + r + 8 * half] : 0.0f;

#pragma unroll
  for (int t = 0; t < 8; ++t) {
    const int col  = jbase + t * 16 + nl;
    const float bj = bias_j ? bias_j[col] : 0.0f;
#pragma unroll
    for (int r = 0; r < 8; ++r) {
      const int row = i0 + r + 8 * half;
      const float v = acc[t][r] + bj + bi[r];
      const size_t idx = strideD * z + (size_t)row * ldD + col;
      if (Df) Df[idx] = v;
      else    Dh[idx] = (bf16)v;
    }
  }
}

// ------------- fused scores + softmax: P[b][n][m] (bf16) -------------------
// Block: 8 waves; block covers 16 rows (n) x all 1024 cols (m).
// Wave w owns 8 m-tiles; full score row block lives in accumulators.
__global__ void attn_softmax(const bf16* __restrict__ theta,  // [B][NP][DIM]
                             const bf16* __restrict__ phi,    // [B][MP][DIM]
                             bf16* __restrict__ P) {          // [B][NP][MP]
  const int b    = blockIdx.y;
  const int n0   = blockIdx.x * 16;
  const int w    = threadIdx.x >> 5;
  const int lane = threadIdx.x & 31;
  const int half = lane >> 4;
  const int nl   = lane & 15;
  const bf16* th = theta + (size_t)b * NP * DIM;
  const bf16* ph = phi   + (size_t)b * MP * DIM;

  floatx8 acc[8];
#pragma unroll
  for (int t = 0; t < 8; ++t) acc[t] = (floatx8)(0.0f);

  for (int k = 0; k < DIM; k += 32) {
    bf16x16 af = load_a_frag(th, DIM, n0, k);
#pragma unroll
    for (int t = 0; t < 8; ++t) {
      const int m0 = (w * 8 + t) * 16;
      bf16x16 bfr = load_b_frag(ph, DIM, m0, k);
      acc[t] = wmma_bf16(af, bfr, acc[t]);
    }
  }

  const float scale = 0.04419417382415922f;  // 512^-0.5
#pragma unroll
  for (int t = 0; t < 8; ++t)
#pragma unroll
    for (int r = 0; r < 8; ++r) acc[t][r] *= scale;

  // --- per-row max: registers -> 16-lane shuffle -> cross-wave LDS ---
  __shared__ float smax[8][16];
  __shared__ float ssum[8][16];

  float rmax[8];
#pragma unroll
  for (int r = 0; r < 8; ++r) {
    float m = -1e30f;
#pragma unroll
    for (int t = 0; t < 8; ++t) m = fmaxf(m, acc[t][r]);
    rmax[r] = m;
  }
#pragma unroll
  for (int msk = 1; msk <= 8; msk <<= 1)
#pragma unroll
    for (int r = 0; r < 8; ++r)
      rmax[r] = fmaxf(rmax[r], __shfl_xor(rmax[r], msk, 32));
  if (nl == 0) {
#pragma unroll
    for (int r = 0; r < 8; ++r) smax[w][8 * half + r] = rmax[r];
  }
  __syncthreads();
#pragma unroll
  for (int r = 0; r < 8; ++r) {
    float m = smax[0][8 * half + r];
#pragma unroll
    for (int w2 = 1; w2 < 8; ++w2) m = fmaxf(m, smax[w2][8 * half + r]);
    rmax[r] = m;
  }

  // --- exp + per-row sum ---
  float rsum[8];
#pragma unroll
  for (int r = 0; r < 8; ++r) rsum[r] = 0.0f;
#pragma unroll
  for (int t = 0; t < 8; ++t)
#pragma unroll
    for (int r = 0; r < 8; ++r) {
      const float e = __expf(acc[t][r] - rmax[r]);
      acc[t][r] = e;
      rsum[r] += e;
    }
#pragma unroll
  for (int msk = 1; msk <= 8; msk <<= 1)
#pragma unroll
    for (int r = 0; r < 8; ++r) rsum[r] += __shfl_xor(rsum[r], msk, 32);
  if (nl == 0) {
#pragma unroll
    for (int r = 0; r < 8; ++r) ssum[w][8 * half + r] = rsum[r];
  }
  __syncthreads();
#pragma unroll
  for (int r = 0; r < 8; ++r) {
    float s = 0.0f;
#pragma unroll
    for (int w2 = 0; w2 < 8; ++w2) s += ssum[w2][8 * half + r];
    rsum[r] = 1.0f / s;
  }

  // --- normalize, store bf16 P ---
  bf16* Pb = P + (size_t)b * NP * MP;
#pragma unroll
  for (int t = 0; t < 8; ++t) {
    const int m0 = (w * 8 + t) * 16;
#pragma unroll
    for (int r = 0; r < 8; ++r) {
      const int row = n0 + r + 8 * half;
      Pb[(size_t)row * MP + m0 + nl] = (bf16)(acc[t][r] * rsum[r]);
    }
  }
}

// ---------------------- elementwise / reshuffle kernels --------------------

__global__ void cvt_f32_bf16(const float* __restrict__ src,
                             bf16* __restrict__ dst, int n) {
  const int i = blockIdx.x * blockDim.x + threadIdx.x;
  if (i < n) dst[i] = (bf16)src[i];
}

// x [b][c][n] f32 -> xb [b][n][c] bf16
__global__ void transpose_cvt_x(const float* __restrict__ x,
                                bf16* __restrict__ xb) {
  const size_t tid = (size_t)blockIdx.x * blockDim.x + threadIdx.x;
  const size_t b   = tid / ((size_t)CINC * NP);
  const size_t rem = tid % ((size_t)CINC * NP);
  const size_t c   = rem / NP;
  const size_t n   = rem % NP;
  xb[(b * NP + n) * CINC + c] = (bf16)x[tid];
}

// maxpool 2x2 + transpose: x [b][c][64][64] -> poolb [b][m][c] bf16
__global__ void pool_cvt(const float* __restrict__ x,
                         bf16* __restrict__ poolb) {
  const size_t tid = (size_t)blockIdx.x * blockDim.x + threadIdx.x;
  const int c = (int)(tid % CINC);
  const int m = (int)((tid / CINC) % MP);
  const int b = (int)(tid / ((size_t)CINC * MP));
  const int my = m >> 5, mx = m & 31;
  const float* px = x + (((size_t)b * CINC + c) * HH + 2 * my) * WWID + 2 * mx;
  const float v = fmaxf(fmaxf(px[0], px[1]), fmaxf(px[WWID], px[WWID + 1]));
  poolb[tid] = (bf16)v;
}

// deterministic per-(b,co) partial sums of y and y^2
__global__ void row_stats(const float* __restrict__ y,
                          float* __restrict__ psum,
                          float* __restrict__ psumsq) {
  const int co = blockIdx.x, b = blockIdx.y, tid = threadIdx.x;
  const float* row = y + ((size_t)b * COUT + co) * NP;
  float s = 0.0f, ss = 0.0f;
  for (int i = tid; i < NP; i += 256) {
    const float v = row[i];
    s += v; ss += v * v;
  }
  __shared__ float rs[256], rss[256];
  rs[tid] = s; rss[tid] = ss;
  __syncthreads();
  for (int st = 128; st > 0; st >>= 1) {
    if (tid < st) { rs[tid] += rs[tid + st]; rss[tid] += rss[tid + st]; }
    __syncthreads();
  }
  if (tid == 0) { psum[co * NB + b] = rs[0]; psumsq[co * NB + b] = rss[0]; }
}

__global__ void finalize_stats(const float* __restrict__ psum,
                               const float* __restrict__ psumsq,
                               float* __restrict__ meanv,
                               float* __restrict__ invstd) {
  const int co = blockIdx.x * blockDim.x + threadIdx.x;
  if (co < COUT) {
    float s = 0.0f, ss = 0.0f;
#pragma unroll
    for (int b = 0; b < NB; ++b) { s += psum[co * NB + b]; ss += psumsq[co * NB + b]; }
    const float cnt = (float)(NB * NP);
    const float m = s / cnt;
    const float var = ss / cnt - m * m;
    meanv[co]  = m;
    invstd[co] = rsqrtf(var + BN_EPS);
  }
}

__global__ void bn_residual(float* __restrict__ y, const float* __restrict__ x,
                            const float* __restrict__ meanv,
                            const float* __restrict__ invstd,
                            const float* __restrict__ bnw,
                            const float* __restrict__ bnb) {
  const size_t idx = (size_t)blockIdx.x * blockDim.x + threadIdx.x;
  const int co = (int)((idx >> 12) & (COUT - 1));  // layout [b][co][4096]
  y[idx] = (y[idx] - meanv[co]) * invstd[co] * bnw[co] + bnb[co] + x[idx];
}

// ------------------------------- launcher ----------------------------------

extern "C" void kernel_launch(void* const* d_in, const int* in_sizes, int n_in,
                              void* d_out, int out_size, void* d_ws, size_t ws_size,
                              hipStream_t stream) {
  (void)in_sizes; (void)n_in; (void)out_size; (void)ws_size;
  const float* x       = (const float*)d_in[0];
  const float* theta_w = (const float*)d_in[1];
  const float* theta_b = (const float*)d_in[2];
  const float* phi_w   = (const float*)d_in[3];
  const float* phi_b   = (const float*)d_in[4];
  const float* g_w     = (const float*)d_in[5];
  const float* g_b     = (const float*)d_in[6];
  const float* out_w   = (const float*)d_in[7];
  const float* out_b   = (const float*)d_in[8];
  const float* bn_w    = (const float*)d_in[9];
  const float* bn_b    = (const float*)d_in[10];
  float* y = (float*)d_out;  // [b][co][n] f32

  char* ws = (char*)d_ws;
  const size_t MB = 1ull << 20;
  bf16* xb     = (bf16*)(ws + 0);          // 64 MB [b][n][c]   (dead after theta)
  bf16* pb     = (bf16*)(ws + 0);          // 64 MB [b][n][m]   (reuses xb space)
  bf16* poolb  = (bf16*)(ws + 64 * MB);    // 16 MB [b][m][c]
  bf16* thetab = (bf16*)(ws + 80 * MB);    // 32 MB [b][n][d]
  bf16* phib   = (bf16*)(ws + 112 * MB);   //  8 MB [b][m][d]
  bf16* gb     = (bf16*)(ws + 120 * MB);   //  8 MB [b][d][m]
  bf16* tb     = (bf16*)(ws + 128 * MB);   // 32 MB [b][n][d]
  bf16* thw    = (bf16*)(ws + 160 * MB);   //  1 MB each weight
  bf16* phw    = (bf16*)(ws + 161 * MB);
  bf16* gw     = (bf16*)(ws + 162 * MB);
  bf16* outw   = (bf16*)(ws + 163 * MB);
  float* psum   = (float*)(ws + 164 * MB); // [COUT][NB]
  float* psumsq = psum + COUT * NB;
  float* meanv  = psumsq + COUT * NB;
  float* invstd = meanv + COUT;

  // weights -> bf16
  cvt_f32_bf16<<<(DIM * CINC) / 256, 256, 0, stream>>>(theta_w, thw, DIM * CINC);
  cvt_f32_bf16<<<(DIM * CINC) / 256, 256, 0, stream>>>(phi_w,   phw, DIM * CINC);
  cvt_f32_bf16<<<(DIM * CINC) / 256, 256, 0, stream>>>(g_w,     gw,  DIM * CINC);
  cvt_f32_bf16<<<(COUT * DIM) / 256, 256, 0, stream>>>(out_w,   outw, COUT * DIM);

  // activations -> bf16, K-contiguous layouts
  transpose_cvt_x<<<(NB * CINC * NP) / 256, 256, 0, stream>>>(x, xb);
  pool_cvt<<<(NB * MP * CINC) / 256, 256, 0, stream>>>(x, poolb);

  // theta^T [b][n][d] = xb [n][c] * thw [d][c] + theta_b[d]   (J=512)
  gemm_nt<<<dim3(NP / 16, DIM / 512, NB), 128, 0, stream>>>(
      xb, (size_t)NP * CINC, CINC, thw, 0, CINC,
      thetab, nullptr, (size_t)NP * DIM, DIM, nullptr, theta_b, CINC);
  // phi^T [b][m][d]                                           (J=512)
  gemm_nt<<<dim3(MP / 16, DIM / 512, NB), 128, 0, stream>>>(
      poolb, (size_t)MP * CINC, CINC, phw, 0, CINC,
      phib, nullptr, (size_t)MP * DIM, DIM, nullptr, phi_b, CINC);
  // g [b][d][m] = gw [d][c] * poolb [m][c] + g_b[d]           (J=1024)
  gemm_nt<<<dim3(DIM / 16, MP / 512, NB), 128, 0, stream>>>(
      gw, 0, CINC, poolb, (size_t)MP * CINC, CINC,
      gb, nullptr, (size_t)DIM * MP, MP, g_b, nullptr, CINC);

  // fused scores + softmax -> P [b][n][m] bf16 (overwrites xb region)
  attn_softmax<<<dim3(NP / 16, NB), 256, 0, stream>>>(thetab, phib, pb);

  // t^T [b][n][d] = P [n][m] * g [d][m]                       (J=512)
  gemm_nt<<<dim3(NP / 16, DIM / 512, NB), 128, 0, stream>>>(
      pb, (size_t)NP * MP, MP, gb, (size_t)DIM * MP, MP,
      tb, nullptr, (size_t)NP * DIM, DIM, nullptr, nullptr, MP);

  // y [b][co][n] f32 = outw [co][d] * t^T [n][d] + out_b[co]  (J=4096)
  gemm_nt<<<dim3(COUT / 16, NP / 512, NB), 128, 0, stream>>>(
      outw, 0, DIM, tb, (size_t)NP * DIM, DIM,
      nullptr, y, (size_t)COUT * NP, NP, out_b, nullptr, DIM);

  // BatchNorm (training stats, deterministic two-stage) + residual
  row_stats<<<dim3(COUT, NB), 256, 0, stream>>>(y, psum, psumsq);
  finalize_stats<<<COUT / 256, 256, 0, stream>>>(psum, psumsq, meanv, invstd);
  bn_residual<<<(size_t)NB * COUT * NP / 256, 256, 0, stream>>>(
      y, x, meanv, invstd, bn_w, bn_b);
}
